// ConcatMLPUpdate_44341242364188
// MI455X (gfx1250) — matrix operand order repaired and checked by reading, not compiled, not run on hardware
//
#include <hip/hip_runtime.h>
#include <stdint.h>

// ---------------------------------------------------------------------------
// Types for CDNA5 WMMA (wave32, v_wmma_f32_16x16x32_bf16)
// ---------------------------------------------------------------------------
typedef __attribute__((ext_vector_type(16))) __bf16 v16bf;
typedef __attribute__((ext_vector_type(8)))  __bf16 v8bf;
typedef __attribute__((ext_vector_type(8)))  float  v8f;

#define DEV __device__ __forceinline__

DEV uint16_t f2bf_u(float f) {
  uint32_t u = __builtin_bit_cast(uint32_t, f);
  uint32_t r = (u + 0x7FFFu + ((u >> 16) & 1u)) >> 16;
  return (uint16_t)r;
}
DEV float bfu2f(uint16_t s) {
  return __builtin_bit_cast(float, (uint32_t)s << 16);
}
DEV float gelu_tanh(float x) {
  float x3 = x * x * x;
  return 0.5f * x * (1.0f + tanhf(0.7978845608028654f * (x + 0.044715f * x3)));
}
DEV v16bf cat16(v8bf a0, v8bf a1) {
  return __builtin_shufflevector(a0, a1, 0, 1, 2, 3, 4, 5, 6, 7,
                                 8, 9, 10, 11, 12, 13, 14, 15);
}

// Dims
#define BB 32
#define TT 512
#define DD 768
#define MM 2048
#define KK 32
#define RR 128
#define HH 1024
#define MK 65536

#define EPI_NONE 0
#define EPI_GELU 1
#define EPI_RES  2

// ---------------------------------------------------------------------------
// Weight packing: W[f32, Kd x N row-major] -> packed bf16 B-fragment layout
// packed[((kb*NT + nb)*32 + lane)*16 + i]:
//   col = nb*16 + (lane & 15);  k = kb*32 + (lane>>4)*8 + (i&7) + (i>>3)*16
// ---------------------------------------------------------------------------
__global__ void pack_w(const float* __restrict__ W, uint16_t* __restrict__ P,
                       int Kd, int N) {
  int o = blockIdx.x * blockDim.x + threadIdx.x;
  if (o >= Kd * N) return;
  int i    = o & 15;
  int lane = (o >> 4) & 31;
  int t    = o >> 9;
  int NT   = N >> 4;
  int nb   = t % NT;
  int kb   = t / NT;
  int col  = nb * 16 + (lane & 15);
  int k    = kb * 32 + (lane >> 4) * 8 + (i & 7) + (i >> 3) * 16;
  P[o] = f2bf_u(W[(size_t)k * N + col]);
}

// ---------------------------------------------------------------------------
// Gather start/end encodings -> Xg bf16 [M, 1536]
// ---------------------------------------------------------------------------
__global__ void gather_concat(const float* __restrict__ enc,
                              const int* __restrict__ bpos,
                              const int* __restrict__ spos,
                              const int* __restrict__ epos,
                              uint16_t* __restrict__ Xg) {
  int tid = blockIdx.x * blockDim.x + threadIdx.x;
  int m = tid / 1536;
  int j = tid - m * 1536;
  if (m >= MM) return;
  int b   = bpos[m];
  int pos = (j < DD) ? spos[m] : epos[m];
  int d   = (j < DD) ? j : j - DD;
  Xg[(size_t)m * 1536 + j] = f2bf_u(enc[((size_t)b * TT + pos) * DD + d]);
}

// ---------------------------------------------------------------------------
// X2 bf16 [MK, 256] = concat(pmv[m] broadcast over K, retrieval_values)
// ---------------------------------------------------------------------------
__global__ void build_x2(const uint16_t* __restrict__ pmv,
                         const float* __restrict__ rv,
                         uint16_t* __restrict__ X2) {
  size_t tid = (size_t)blockIdx.x * blockDim.x + threadIdx.x;
  size_t row = tid >> 8;
  int    j   = (int)(tid & 255);
  if (row >= (size_t)MK) return;
  int m = (int)(row >> 5);
  uint16_t v = (j < RR) ? pmv[(size_t)m * RR + j]
                        : f2bf_u(rv[row * RR + (j - RR)]);
  X2[tid] = v;
}

// ---------------------------------------------------------------------------
// Epilogue helper shared by both GEMM variants
// ---------------------------------------------------------------------------
template <int NN, int EPI>
DEV void store_tile(const v8f acc[4][2], const uint16_t* __restrict__ Res,
                    uint16_t* __restrict__ Out, int row0, int col0, int l,
                    int hi) {
#pragma unroll
  for (int r = 0; r < 4; ++r) {
#pragma unroll
    for (int c = 0; c < 2; ++c) {
      const size_t obase =
          (size_t)(row0 + r * 16 + hi * 8) * NN + col0 + c * 16 + l;
#pragma unroll
      for (int e = 0; e < 8; ++e) {
        float v = acc[r][c][e];
        if (EPI == EPI_GELU) v = gelu_tanh(v);
        if (EPI == EPI_RES)  v += bfu2f(Res[obase + (size_t)e * NN]);
        Out[obase + (size_t)e * NN] = f2bf_u(v);
      }
    }
  }
}

// ---------------------------------------------------------------------------
// LDS-staged WMMA GEMM (requires NN % 256 == 0, rows % 64 == 0).
// Block (8 waves) owns a 64x256 output tile; the shared 64x32 A tile is
// double-buffered through LDS (stride 40 -> conflict-free b128 frag reads).
// Each wave keeps a 64x32 register tile: 4 A-frags x 2 B-frags -> 8 WMMAs/step.
// ---------------------------------------------------------------------------
template <int KD, int NN, int EPI>
__global__ void __launch_bounds__(256)
gemm_bf16_lds(const uint16_t* __restrict__ A, const uint16_t* __restrict__ Wp,
              const float* __restrict__ bias, const uint16_t* __restrict__ Res,
              uint16_t* __restrict__ Out, int rows) {
  constexpr int NT = NN / 16;
  constexpr int CB = NN / 256;           // 256-wide col blocks
  constexpr int NKB = KD / 32;
  constexpr int LDA = 40;                // padded LDS row stride (elements)

  const int rowBlk = blockIdx.x / CB;
  const int colBlk = blockIdx.x - rowBlk * CB;
  const int wave = threadIdx.x >> 5;
  const int lane = threadIdx.x & 31;
  const int l  = lane & 15;
  const int hi = lane >> 4;
  const int row0 = rowBlk << 6;
  const int col0 = (colBlk << 8) + (wave << 5);  // wave's 32-col macro-tile

  __shared__ alignas(16) uint16_t smA[2][64 * LDA];

  v8f acc[4][2];
#pragma unroll
  for (int c = 0; c < 2; ++c) {
    const float bv = bias[col0 + c * 16 + l];
#pragma unroll
    for (int r = 0; r < 4; ++r)
      acc[r][c] = (v8f){bv, bv, bv, bv, bv, bv, bv, bv};
  }

  // Cooperative A staging: thread t loads one 16B chunk of the 64x32 tile.
  const int tr = threadIdx.x >> 2;        // 0..63 (row in tile)
  const int tc = (threadIdx.x & 3) * 8;   // 0,8,16,24 (k offset)
  const uint16_t* __restrict__ gA = A + (size_t)(row0 + tr) * KD + tc;

  // B: packed; wave's two 16-col subtiles are 512 elements apart per kb.
  const int nb0 = (col0 >> 4);
  const uint16_t* __restrict__ bp = Wp + ((size_t)nb0 * 32 + lane) * 16;
  constexpr size_t BSTEP = (size_t)NT * 512;

  // preload kb = 0
  {
    v8bf v = *(const v8bf*)gA;
    *(v8bf*)(&smA[0][tr * LDA + tc]) = v;
  }
  __syncthreads();

#pragma unroll 2
  for (int kb = 0; kb < NKB; ++kb) {
    const int cur = kb & 1;
    // stage next A tile into the other buffer while computing this one
    if (kb + 1 < NKB) {
      v8bf v = *(const v8bf*)(gA + (kb + 1) * 32);
      *(v8bf*)(&smA[cur ^ 1][tr * LDA + tc]) = v;
    }
    // A fragments from LDS (conflict-free b128 reads)
    v16bf af[4];
#pragma unroll
    for (int r = 0; r < 4; ++r) {
      const uint16_t* lp = &smA[cur][(r * 16 + l) * LDA + hi * 8];
      af[r] = cat16(*(const v8bf*)lp, *(const v8bf*)(lp + 16));
    }
    // B fragments from global (L2-resident packed weights)
    v16bf bf[2];
#pragma unroll
    for (int c = 0; c < 2; ++c)
      bf[c] = *(const v16bf*)(bp + c * 512);
    __builtin_prefetch(bp + BSTEP, 0, 1);
    bp += BSTEP;
#pragma unroll
    for (int r = 0; r < 4; ++r)
#pragma unroll
      for (int c = 0; c < 2; ++c)
        acc[r][c] = __builtin_amdgcn_wmma_f32_16x16x32_bf16(
            false, af[r], false, bf[c], (short)0, acc[r][c], false, false);
    __syncthreads();
  }

  store_tile<NN, EPI>(acc, Res, Out, row0, col0, l, hi);
}

// ---------------------------------------------------------------------------
// Direct register-blocked WMMA GEMM (for small N, e.g. pmv N=128)
// ---------------------------------------------------------------------------
template <int KD, int NN, int EPI>
__global__ void __launch_bounds__(256)
gemm_bf16_direct(const uint16_t* __restrict__ A,
                 const uint16_t* __restrict__ Wp,
                 const float* __restrict__ bias,
                 const uint16_t* __restrict__ Res,
                 uint16_t* __restrict__ Out, int rows) {
  constexpr int NT = NN / 16;
  constexpr int CT = NN / 32;
  const int tile = blockIdx.x * 8 + (threadIdx.x >> 5);
  const int totalTiles = (rows >> 6) * CT;
  if (tile >= totalTiles) return;

  const int rowTile = tile / CT;
  const int colTile = tile - rowTile * CT;
  const int row0 = rowTile << 6;
  const int col0 = colTile << 5;
  const int lane = threadIdx.x & 31;
  const int l  = lane & 15;
  const int hi = lane >> 4;

  v8f acc[4][2];
#pragma unroll
  for (int c = 0; c < 2; ++c) {
    const float bv = bias[col0 + c * 16 + l];
#pragma unroll
    for (int r = 0; r < 4; ++r)
      acc[r][c] = (v8f){bv, bv, bv, bv, bv, bv, bv, bv};
  }

  const uint16_t* __restrict__ arow[4];
#pragma unroll
  for (int r = 0; r < 4; ++r)
    arow[r] = A + (size_t)(row0 + r * 16 + l) * KD + hi * 8;
  const uint16_t* __restrict__ bp =
      Wp + ((size_t)colTile * 2 * 32 + lane) * 16;
  constexpr size_t BSTEP = (size_t)NT * 512;

#pragma unroll 2
  for (int kb = 0; kb < KD / 32; ++kb) {
    v16bf af[4];
#pragma unroll
    for (int r = 0; r < 4; ++r)
      af[r] = cat16(*(const v8bf*)(arow[r] + kb * 32),
                    *(const v8bf*)(arow[r] + kb * 32 + 16));
    v16bf bf[2];
#pragma unroll
    for (int c = 0; c < 2; ++c)
      bf[c] = *(const v16bf*)(bp + c * 512);
    __builtin_prefetch(bp + BSTEP, 0, 1);
    bp += BSTEP;
#pragma unroll
    for (int r = 0; r < 4; ++r)
#pragma unroll
      for (int c = 0; c < 2; ++c)
        acc[r][c] = __builtin_amdgcn_wmma_f32_16x16x32_bf16(
            false, af[r], false, bf[c], (short)0, acc[r][c], false, false);
  }

  store_tile<NN, EPI>(acc, Res, Out, row0, col0, l, hi);
}

// ---------------------------------------------------------------------------
// Row LayerNorm (bf16, in place), one block per row of NC cols
// ---------------------------------------------------------------------------
template <int NC>
__global__ void __launch_bounds__(256)
ln_rows_bf16(uint16_t* __restrict__ buf, const float* __restrict__ sc,
             const float* __restrict__ bi) {
  uint16_t* p = buf + (size_t)blockIdx.x * NC;
  float x[NC / 256];
  float s = 0.f, s2 = 0.f;
#pragma unroll
  for (int i = 0; i < NC / 256; ++i) {
    x[i] = bfu2f(p[threadIdx.x + i * 256]);
    s += x[i];
    s2 += x[i] * x[i];
  }
  __shared__ float rs[256], rs2[256];
  rs[threadIdx.x] = s;
  rs2[threadIdx.x] = s2;
  __syncthreads();
  for (int off = 128; off > 0; off >>= 1) {
    if (threadIdx.x < off) {
      rs[threadIdx.x] += rs[threadIdx.x + off];
      rs2[threadIdx.x] += rs2[threadIdx.x + off];
    }
    __syncthreads();
  }
  float mean = rs[0] / NC;
  float var  = rs2[0] / NC - mean * mean;
  float inv  = rsqrtf(var + 1e-12f);
#pragma unroll
  for (int i = 0; i < NC / 256; ++i) {
    int d = threadIdx.x + i * 256;
    p[d] = f2bf_u((x[i] - mean) * inv * sc[d] + bi[d]);
  }
}

template <int NC>
__global__ void __launch_bounds__(256)
ln_rows_f32(float* __restrict__ buf, const float* __restrict__ sc,
            const float* __restrict__ bi) {
  float* p = buf + (size_t)blockIdx.x * NC;
  float x[NC / 256];
  float s = 0.f, s2 = 0.f;
#pragma unroll
  for (int i = 0; i < NC / 256; ++i) {
    x[i] = p[threadIdx.x + i * 256];
    s += x[i];
    s2 += x[i] * x[i];
  }
  __shared__ float rs[256], rs2[256];
  rs[threadIdx.x] = s;
  rs2[threadIdx.x] = s2;
  __syncthreads();
  for (int off = 128; off > 0; off >>= 1) {
    if (threadIdx.x < off) {
      rs[threadIdx.x] += rs[threadIdx.x + off];
      rs2[threadIdx.x] += rs2[threadIdx.x + off];
    }
    __syncthreads();
  }
  float mean = rs[0] / NC;
  float var  = rs2[0] / NC - mean * mean;
  float inv  = rsqrtf(var + 1e-12f);
#pragma unroll
  for (int i = 0; i < NC / 256; ++i) {
    int d = threadIdx.x + i * 256;
    p[d] = (x[i] - mean) * inv * sc[d] + bi[d];
  }
}

// ---------------------------------------------------------------------------
// pooled[m,:] = sum_k scores[m,k] * z[m,k,:]  (z bf16 post-LN), out bf16
// ---------------------------------------------------------------------------
__global__ void __launch_bounds__(256)
pool_scores(const uint16_t* __restrict__ z, const float* __restrict__ scores,
            uint16_t* __restrict__ pooledb) {
  const int m = blockIdx.x;
  const float* sc = scores + (size_t)m * KK;
  for (int d = threadIdx.x; d < DD; d += 256) {
    float acc = 0.f;
#pragma unroll 8
    for (int k = 0; k < KK; ++k)
      acc += sc[k] * bfu2f(z[((size_t)m * KK + k) * DD + d]);
    pooledb[(size_t)m * DD + d] = f2bf_u(acc);
  }
}

// ---------------------------------------------------------------------------
// Scatter-add masked pooled rows into out at (batch, start)
// ---------------------------------------------------------------------------
__global__ void __launch_bounds__(256)
scatter_add(const uint16_t* __restrict__ pz, const float* __restrict__ mask,
            const int* __restrict__ bpos, const int* __restrict__ spos,
            float* __restrict__ out) {
  const int m = blockIdx.x;
  const float mk = mask[m];
  if (mk == 0.f) return;
  float* dst = out + ((size_t)bpos[m] * TT + spos[m]) * DD;
  const uint16_t* src = pz + (size_t)m * DD;
  for (int d = threadIdx.x; d < DD; d += 256)
    atomicAdd(dst + d, mk * bfu2f(src[d]));
}

// ---------------------------------------------------------------------------
extern "C" void kernel_launch(void* const* d_in, const int* in_sizes, int n_in,
                              void* d_out, int out_size, void* d_ws,
                              size_t ws_size, hipStream_t stream) {
  const float* enc   = (const float*)d_in[0];
  const float* rv    = (const float*)d_in[1];
  const float* rsc   = (const float*)d_in[2];
  const int*   mbp   = (const int*)d_in[3];
  const int*   msp   = (const int*)d_in[4];
  const int*   mep   = (const int*)d_in[5];
  const float* mmask = (const float*)d_in[6];
  const float* vp_w  = (const float*)d_in[7];
  const float* vp_b  = (const float*)d_in[8];
  const float* cm_w  = (const float*)d_in[9];
  const float* cm_b  = (const float*)d_in[10];
  const float* cd_w  = (const float*)d_in[11];
  const float* cd_b  = (const float*)d_in[12];
  const float* ac_w1 = (const float*)d_in[13];
  const float* ac_b1 = (const float*)d_in[14];
  const float* ac_w2 = (const float*)d_in[15];
  const float* ac_b2 = (const float*)d_in[16];
  const float* ac_ls = (const float*)d_in[17];
  const float* ac_lb = (const float*)d_in[18];
  const float* pl_w1 = (const float*)d_in[19];
  const float* pl_b1 = (const float*)d_in[20];
  const float* pl_w2 = (const float*)d_in[21];
  const float* pl_b2 = (const float*)d_in[22];
  const float* pl_ls = (const float*)d_in[23];
  const float* pl_lb = (const float*)d_in[24];
  const float* ln_s  = (const float*)d_in[25];
  const float* ln_b  = (const float*)d_in[26];
  float* out = (float*)d_out;

  char* ws = (char*)d_ws;
  size_t off = 0;
  auto alloc = [&](size_t bytes) -> void* {
    void* p = ws + off;
    off += (bytes + 255) & ~(size_t)255;
    return p;
  };

  uint16_t* wp_vp  = (uint16_t*)alloc((size_t)1536 * 128 * 2);
  uint16_t* wp_cm  = (uint16_t*)alloc((size_t)256 * 1024 * 2);
  uint16_t* wp_cd  = (uint16_t*)alloc((size_t)1024 * 768 * 2);
  uint16_t* wp_ac1 = (uint16_t*)alloc((size_t)768 * 1024 * 2);
  uint16_t* wp_ac2 = (uint16_t*)alloc((size_t)1024 * 768 * 2);
  uint16_t* wp_pl1 = (uint16_t*)alloc((size_t)768 * 1024 * 2);
  uint16_t* wp_pl2 = (uint16_t*)alloc((size_t)1024 * 768 * 2);
  uint16_t* Xg  = (uint16_t*)alloc((size_t)MM * 1536 * 2);
  uint16_t* pmv = (uint16_t*)alloc((size_t)MM * RR * 2);
  uint16_t* X2  = (uint16_t*)alloc((size_t)MK * 256 * 2);
  uint16_t* h1  = (uint16_t*)alloc((size_t)MK * HH * 2);   // reused for h2
  uint16_t* yb  = (uint16_t*)alloc((size_t)MK * DD * 2);
  uint16_t* zb  = (uint16_t*)alloc((size_t)MK * DD * 2);
  uint16_t* plb = (uint16_t*)alloc((size_t)MM * DD * 2);
  uint16_t* ph  = (uint16_t*)alloc((size_t)MM * HH * 2);
  uint16_t* pz  = (uint16_t*)alloc((size_t)MM * DD * 2);
  (void)in_sizes; (void)n_in; (void)out_size; (void)ws_size;

  dim3 blk(256);

  // ---- pack weights into WMMA B-fragment layout (bf16) ----
  pack_w<<<(1536 * 128 + 255) / 256, blk, 0, stream>>>(vp_w, wp_vp, 1536, 128);
  pack_w<<<(256 * 1024 + 255) / 256, blk, 0, stream>>>(cm_w, wp_cm, 256, 1024);
  pack_w<<<(1024 * 768 + 255) / 256, blk, 0, stream>>>(cd_w, wp_cd, 1024, 768);
  pack_w<<<(768 * 1024 + 255) / 256, blk, 0, stream>>>(ac_w1, wp_ac1, 768, 1024);
  pack_w<<<(1024 * 768 + 255) / 256, blk, 0, stream>>>(ac_w2, wp_ac2, 1024, 768);
  pack_w<<<(768 * 1024 + 255) / 256, blk, 0, stream>>>(pl_w1, wp_pl1, 768, 1024);
  pack_w<<<(1024 * 768 + 255) / 256, blk, 0, stream>>>(pl_w2, wp_pl2, 1024, 768);

  // ---- gather + pmv GEMM [2048,1536]x[1536,128] (direct path, N=128) ----
  gather_concat<<<(MM * 1536) / 256, blk, 0, stream>>>(enc, mbp, msp, mep, Xg);
  {
    int tiles = (MM / 64) * (128 / 32);
    gemm_bf16_direct<1536, 128, EPI_NONE><<<(tiles + 7) / 8, blk, 0, stream>>>(
        Xg, wp_vp, vp_b, nullptr, pmv, MM);
  }

  // ---- concat(pmv, rv) -> big GEMM chain over 65536 rows (LDS path) ----
  build_x2<<<((size_t)MK * 256) / 256, blk, 0, stream>>>(pmv, rv, X2);
  gemm_bf16_lds<256, 1024, EPI_GELU>
      <<<(MK / 64) * (HH / 256), blk, 0, stream>>>(X2, wp_cm, cm_b, nullptr,
                                                   h1, MK);
  gemm_bf16_lds<1024, 768, EPI_NONE>
      <<<(MK / 64) * (DD / 256), blk, 0, stream>>>(h1, wp_cd, cd_b, nullptr,
                                                   yb, MK);
  gemm_bf16_lds<768, 1024, EPI_GELU>
      <<<(MK / 64) * (HH / 256), blk, 0, stream>>>(yb, wp_ac1, ac_b1, nullptr,
                                                   h1, MK);
  gemm_bf16_lds<1024, 768, EPI_RES>
      <<<(MK / 64) * (DD / 256), blk, 0, stream>>>(h1, wp_ac2, ac_b2, yb, zb,
                                                   MK);
  ln_rows_bf16<768><<<MK, blk, 0, stream>>>(zb, ac_ls, ac_lb);

  // ---- score pooling + pooled MLP block ----
  pool_scores<<<MM, blk, 0, stream>>>(zb, rsc, plb);
  gemm_bf16_lds<768, 1024, EPI_GELU>
      <<<(MM / 64) * (HH / 256), blk, 0, stream>>>(plb, wp_pl1, pl_b1, nullptr,
                                                   ph, MM);
  gemm_bf16_lds<1024, 768, EPI_RES>
      <<<(MM / 64) * (DD / 256), blk, 0, stream>>>(ph, wp_pl2, pl_b2, plb, pz,
                                                   MM);
  ln_rows_bf16<768><<<MM, blk, 0, stream>>>(pz, pl_ls, pl_lb);

  // ---- out = encoded_input; scatter-add masked pooled; final LN ----
  hipMemcpyAsync(out, enc, (size_t)BB * TT * DD * sizeof(float),
                 hipMemcpyDeviceToDevice, stream);
  scatter_add<<<MM, blk, 0, stream>>>(pz, mmask, mbp, msp, out);
  ln_rows_f32<768><<<BB * TT, blk, 0, stream>>>(out, ln_s, ln_b);
}
